// SimpleMLPWithWrapper_1305670058602
// MI455X (gfx1250) — matrix operand order repaired and checked by reading, not compiled, not run on hardware
//
#include <hip/hip_runtime.h>

// Problem dims (fixed by reference)
#define T_DIM   4096
#define H_DIM   4096
#define I_DIM   14336
#define GROUP   128
#define FP8MAX  448.0f
#define EPS_AMX 1e-10f

// Tile config
#define BM 128
#define BN 128
#define BK 32
#define LDSW 40            // BK + 8 pad (80 B rows: staggers banks, keeps 16B align)
#define NTHREADS 512       // 16 waves (wave32)

typedef __attribute__((ext_vector_type(16))) __bf16 v16bf;
typedef __attribute__((ext_vector_type(8)))  __bf16 v8bf;
typedef __attribute__((ext_vector_type(8)))  float  v8f;

__device__ __forceinline__ v8bf cvt8(const float4 a, const float4 b) {
    v8bf r;
    r[0] = (__bf16)a.x; r[1] = (__bf16)a.y; r[2] = (__bf16)a.z; r[3] = (__bf16)a.w;
    r[4] = (__bf16)b.x; r[5] = (__bf16)b.y; r[6] = (__bf16)b.z; r[7] = (__bf16)b.w;
    return r;
}

__device__ __forceinline__ v16bf combine16(v8bf lo, v8bf hi) {
    v16bf r;
#pragma unroll
    for (int i = 0; i < 8; ++i) { r[i] = lo[i]; r[i + 8] = hi[i]; }
    return r;
}

// Per-thread register staging of one 128x32 fp32 tile slice (x, gateW, upW)
struct Stage { float4 a0, a1, g0, g1, u0, u1; };

__device__ __forceinline__ Stage stage_load(const float* __restrict__ x,
                                            const float* __restrict__ g,
                                            const float* __restrict__ u, int k) {
    Stage s;
    s.a0 = *(const float4*)(x + k); s.a1 = *(const float4*)(x + k + 4);
    s.g0 = *(const float4*)(g + k); s.g1 = *(const float4*)(g + k + 4);
    s.u0 = *(const float4*)(u + k); s.u1 = *(const float4*)(u + k + 4);
    return s;
}

__device__ __forceinline__ void stage_store(const Stage& s,
                                            __bf16 (*As)[LDSW], __bf16 (*Bg)[LDSW],
                                            __bf16 (*Bu)[LDSW], int lr, int lc) {
    *(v8bf*)&As[lr][lc] = cvt8(s.a0, s.a1);
    *(v8bf*)&Bg[lr][lc] = cvt8(s.g0, s.g1);
    *(v8bf*)&Bu[lr][lc] = cvt8(s.u0, s.u1);
}

// One K-step of WMMA math out of a given LDS stage buffer.
__device__ __forceinline__ void wmma_step(const __bf16 (*As)[LDSW],
                                          const __bf16 (*Bg)[LDSW],
                                          const __bf16 (*Bu)[LDSW],
                                          int wm, int wn, int half, int l16,
                                          v8f cg[2][2], v8f cu[2][2])
{
    // A (16x32 bf16): lane l16 = row M, contiguous run K=[16h,16h+16)
    v16bf af[2];
#pragma unroll
    for (int tm = 0; tm < 2; ++tm) {
        const int m  = wm * 32 + tm * 16 + l16;
        const int kb = half * 16;
        af[tm] = combine16(*(const v8bf*)&As[m][kb],
                           *(const v8bf*)&As[m][kb + 8]);
    }
    // B (32x16 bf16): lane l16 = col N, contiguous run K=[16h,16h+16)
    v16bf bgf[2], buf2[2];
#pragma unroll
    for (int tn = 0; tn < 2; ++tn) {
        const int n  = wn * 32 + tn * 16 + l16;
        const int kb = half * 16;
        bgf[tn]  = combine16(*(const v8bf*)&Bg[n][kb],
                             *(const v8bf*)&Bg[n][kb + 8]);
        buf2[tn] = combine16(*(const v8bf*)&Bu[n][kb],
                             *(const v8bf*)&Bu[n][kb + 8]);
    }
#pragma unroll
    for (int tm = 0; tm < 2; ++tm)
#pragma unroll
        for (int tn = 0; tn < 2; ++tn) {
            cg[tm][tn] = __builtin_amdgcn_wmma_f32_16x16x32_bf16(
                false, af[tm], false, bgf[tn], (short)0, cg[tm][tn], false, false);
            cu[tm][tn] = __builtin_amdgcn_wmma_f32_16x16x32_bf16(
                false, af[tm], false, buf2[tn], (short)0, cu[tm][tn], false, false);
        }
}

__global__ __launch_bounds__(NTHREADS)
void fused_gateup_silu_fp8q(const float* __restrict__ X,
                            const float* __restrict__ W,
                            float* __restrict__ outQ,
                            float* __restrict__ outS)
{
    // Double-buffered bf16 stages
    __shared__ __bf16 As[2][BM][LDSW];
    __shared__ __bf16 Bg[2][BN][LDSW];
    __shared__ __bf16 Bu[2][BN][LDSW];
    __shared__ float  P[BM][4];       // per-row partial amax, one slot per wave_n

    const int tid  = threadIdx.x;
    const int lane = tid & 31;
    const int wave = tid >> 5;
    const int wm   = wave >> 2;       // 0..3 : 32-row slice of block tile
    const int wn   = wave & 3;        // 0..3 : 32-col slice of block tile
    const int half = lane >> 4;       // WMMA half-wave
    const int l16  = lane & 15;

    const int row0 = blockIdx.x * BM;     // rows of x / output
    const int col0 = blockIdx.y * BN;     // y columns; gate row = col0+n, up row = I+col0+n

    // ---- global->LDS staging indices: 512 threads * 8 f32 = 128x32 tile ----
    const int lr = tid >> 2;              // 0..127
    const int lc = (tid & 3) * 8;         // 0,8,16,24

    const float* xptr = X + (size_t)(row0 + lr) * H_DIM + lc;
    const float* gptr = W + (size_t)(col0 + lr) * H_DIM + lc;
    const float* uptr = W + (size_t)(I_DIM + col0 + lr) * H_DIM + lc;

    v8f cg[2][2], cu[2][2];
#pragma unroll
    for (int a = 0; a < 2; ++a)
#pragma unroll
        for (int b = 0; b < 2; ++b) { cg[a][b] = (v8f){}; cu[a][b] = (v8f){}; }

    // ---------------- prologue: fill stage 0 with K-step 0 ----------------
    {
        Stage s0 = stage_load(xptr, gptr, uptr, 0);
        stage_store(s0, As[0], Bg[0], Bu[0], lr, lc);
        __builtin_prefetch(xptr + BK, 0, 1);
        __builtin_prefetch(gptr + BK, 0, 1);
        __builtin_prefetch(uptr + BK, 0, 1);
    }
    __syncthreads();

    // ------- pipelined main loop: 2 K-steps/iter, last pair peeled -------
    for (int k0 = 0; k0 < H_DIM - 2 * BK; k0 += 2 * BK) {
        // step A: issue loads for k0+BK, compute stage 0, fill stage 1
        {
            Stage s = stage_load(xptr, gptr, uptr, k0 + BK);
            __builtin_prefetch(xptr + k0 + 2 * BK, 0, 1);
            __builtin_prefetch(gptr + k0 + 2 * BK, 0, 1);
            __builtin_prefetch(uptr + k0 + 2 * BK, 0, 1);
            wmma_step(As[0], Bg[0], Bu[0], wm, wn, half, l16, cg, cu);
            stage_store(s, As[1], Bg[1], Bu[1], lr, lc);
        }
        __syncthreads();
        // step B: issue loads for k0+2*BK, compute stage 1, fill stage 0
        {
            Stage s = stage_load(xptr, gptr, uptr, k0 + 2 * BK);
            __builtin_prefetch(xptr + k0 + 3 * BK, 0, 1);
            __builtin_prefetch(gptr + k0 + 3 * BK, 0, 1);
            __builtin_prefetch(uptr + k0 + 3 * BK, 0, 1);
            wmma_step(As[1], Bg[1], Bu[1], wm, wn, half, l16, cg, cu);
            stage_store(s, As[0], Bg[0], Bu[0], lr, lc);
        }
        __syncthreads();
    }
    // ------------------------ peeled tail pair ------------------------
    {
        Stage s = stage_load(xptr, gptr, uptr, H_DIM - BK);
        wmma_step(As[0], Bg[0], Bu[0], wm, wn, half, l16, cg, cu);
        stage_store(s, As[1], Bg[1], Bu[1], lr, lc);
    }
    __syncthreads();
    wmma_step(As[1], Bg[1], Bu[1], wm, wn, half, l16, cg, cu);

    // ---- epilogue: y = silu(gate) * up, reuse cg as y ----
#pragma unroll
    for (int tm = 0; tm < 2; ++tm)
#pragma unroll
        for (int tn = 0; tn < 2; ++tn)
#pragma unroll
            for (int r = 0; r < 8; ++r) {
                const float g = cg[tm][tn][r];
                const float u = cu[tm][tn][r];
                cg[tm][tn][r] = (g / (1.0f + __expf(-g))) * u;
            }

    // ---- per-row amax over this wave's 32 columns (half-wave shfl reduce) ----
#pragma unroll
    for (int tm = 0; tm < 2; ++tm)
#pragma unroll
        for (int r = 0; r < 8; ++r) {
            float v = fmaxf(fabsf(cg[tm][0][r]), fabsf(cg[tm][1][r]));
            v = fmaxf(v, __shfl_xor(v, 1, 32));
            v = fmaxf(v, __shfl_xor(v, 2, 32));
            v = fmaxf(v, __shfl_xor(v, 4, 32));
            v = fmaxf(v, __shfl_xor(v, 8, 32));
            const int m = wm * 32 + tm * 16 + half * 8 + r;   // C layout: M = 8*half + r
            if (l16 == 0) P[m][wn] = v;
        }
    __syncthreads();

    // ---- group scale (GROUP == BN: one group per row per block), quantize, store ----
    const int scols = I_DIM / GROUP;   // 112
#pragma unroll
    for (int tm = 0; tm < 2; ++tm)
#pragma unroll
        for (int r = 0; r < 8; ++r) {
            const int m = wm * 32 + tm * 16 + half * 8 + r;
            float amax = fmaxf(fmaxf(P[m][0], P[m][1]), fmaxf(P[m][2], P[m][3]));
            amax = fmaxf(amax, EPS_AMX);
            const float scale = amax * (1.0f / FP8MAX);
            const float inv   = FP8MAX / amax;
#pragma unroll
            for (int tn = 0; tn < 2; ++tn) {
                float q = cg[tm][tn][r] * inv;
                q = fminf(fmaxf(q, -FP8MAX), FP8MAX);
                // round through e4m3 so stored float == fp8 value
                const int pk = __builtin_amdgcn_cvt_pk_fp8_f32(q, q, 0, false);
                const float qv = __builtin_amdgcn_cvt_f32_fp8(pk, 0);
                outQ[(size_t)(row0 + m) * I_DIM + col0 + wn * 32 + tn * 16 + l16] = qv;
            }
            if (wn == 0 && l16 == 0)
                outS[(size_t)(row0 + m) * scols + blockIdx.y] = scale;
        }
}

extern "C" void kernel_launch(void* const* d_in, const int* in_sizes, int n_in,
                              void* d_out, int out_size, void* d_ws, size_t ws_size,
                              hipStream_t stream) {
    (void)in_sizes; (void)n_in; (void)out_size; (void)d_ws; (void)ws_size;
    const float* x = (const float*)d_in[0];
    const float* W = (const float*)d_in[1];
    float* outQ = (float*)d_out;                         // [T, I] fp8-rounded values
    float* outS = outQ + (size_t)T_DIM * I_DIM;          // [T, I/128] scales
    dim3 grid(T_DIM / BM, I_DIM / BN);                   // 32 x 112
    fused_gateup_silu_fp8q<<<grid, NTHREADS, 0, stream>>>(x, W, outQ, outS);
}
